// CheckpointedIntegratedMTAPFM_neighbours_24575802867927
// MI455X (gfx1250) — matrix operand order repaired and checked by reading, not compiled, not run on hardware
//
#include <hip/hip_runtime.h>

typedef __attribute__((ext_vector_type(16))) _Float16     v16h;
typedef __attribute__((ext_vector_type(8)))  float        v8f;
typedef __attribute__((ext_vector_type(4)))  unsigned int u32x4;
typedef __attribute__((ext_vector_type(4)))  float        f32x4;

#define DEVINL __device__ __forceinline__
#define AS1 __attribute__((address_space(1)))

// global-address-space pointee types (so laundered pointers still codegen to
// global_load_* with SGPR base instead of flat_load_*)
typedef AS1 const _Float16 GF16;
typedef AS1 const float    GF32;
typedef AS1 const u32x4    GU4;
typedef AS1 const f32x4    GFl4;

union Frag { v16h h; unsigned int u[8]; };

// ---------------- helpers ----------------

DEVINL unsigned int pack2h(float a, float b) {
  _Float16 ha = (_Float16)a, hb = (_Float16)b;
  unsigned short ua = __builtin_bit_cast(unsigned short, ha);
  unsigned short ub = __builtin_bit_cast(unsigned short, hb);
  return (unsigned int)ua | ((unsigned int)ub << 16);
}
DEVINL float h2f_lo(unsigned int u) {
  return (float)__builtin_bit_cast(_Float16, (unsigned short)(u & 0xffffu));
}
DEVINL float h2f_hi(unsigned int u) {
  return (float)__builtin_bit_cast(_Float16, (unsigned short)(u >> 16));
}

DEVINL v8f wmma16(v16h a, v16h b, v8f c) {
  // D = A(16x32 f16) x B(32x16 f16) + C(16x16 f32)
  return __builtin_amdgcn_wmma_f32_16x16x32_f16(false, a, false, b, (short)0, c,
                                                false, false);
}

// Defeat LICM (prevents mass-hoist + scratch spilling) and pin the base in an
// SGPR pair; re-cast to addrspace(1) so loads stay on the GLOBAL path.
DEVINL GF16* launder_h(const _Float16* p) {
  unsigned long long v = (unsigned long long)p;
  asm volatile("" : "+s"(v));
  return (GF16*)v;
}
DEVINL GF32* launder_f(const float* p) {
  unsigned long long v = (unsigned long long)p;
  asm volatile("" : "+s"(v));
  return (GF32*)v;
}

// CDNA5 hardware tanh (single TRANS op) + tanh-based logistic sigmoid.
#if __has_builtin(__builtin_amdgcn_tanhf)
DEVINL float htanh(float x) { return __builtin_amdgcn_tanhf(x); }
#else
DEVINL float htanh(float x) {
  float r;
  asm("v_tanh_f32 %0, %1\n\tv_nop" : "=v"(r) : "v"(x));
  return r;
}
#endif
DEVINL float sigmf(float x) { return fmaf(htanh(0.5f * x), 0.5f, 0.5f); }

#if __has_builtin(__builtin_amdgcn_rcpf)
DEVINL float hrcp(float x) { return __builtin_amdgcn_rcpf(x); }
#else
DEVINL float hrcp(float x) { return 1.0f / x; }
#endif

// A-fragment (16x32, f16): lane<16: M=lane, K={base..base+15},
// lane>=16: M=lane-16, base+=16.  p points at W[m*64 + kt*32 + 16*hi].
DEVINL v16h load_afrag(GF16* p) {
  Frag f;
  u32x4 a = *(GU4*)p;
  u32x4 b = *(GU4*)(p + 8);
  f.u[0] = a[0]; f.u[1] = a[1]; f.u[2] = a[2]; f.u[3] = a[3];
  f.u[4] = b[0]; f.u[5] = b[1]; f.u[6] = b[2]; f.u[7] = b[3];
  return f.h;
}

// load the 4 A-fragments (Wih kt0/kt1, Whh kt0/kt1) of one gate tile
DEVINL void load_gate_frags(GF16* Wih, GF16* Whh, int mt, int r, int hi,
                            v16h* fr) {
  const int row = (mt * 16 + r) * 64 + 16 * hi;
  fr[0] = load_afrag(Wih + row);
  fr[1] = load_afrag(Wih + row + 32);
  fr[2] = load_afrag(Whh + row);
  fr[3] = load_afrag(Whh + row + 32);
}

// ---------------- LSTM cell: 2 sequence-tiles share every A-fragment --------

struct LayerState2 {
  v16h hf[2][2];  // [tile][kt] h^T as B-fragments (K=hidden feat, N=sequence)
  v8f  c[2][4];   // [tile][chunk] cell state in C/D layout
};

DEVINL void lstm_cell2(const _Float16* __restrict__ Wih_,
                       const _Float16* __restrict__ Whh_,
                       const float* __restrict__ bias_,
                       const v16h xf[2][2], LayerState2& st, int lane) {
  GF16* Wih = launder_h(Wih_);
  GF16* Whh = launder_h(Whh_);
  GF32* bias = launder_f(bias_);
  const int r  = lane & 15;
  const int hi = lane >> 4;
  unsigned int hpk[2][4][4];

  // software pipeline: prefetch gate tile 0 (k=0, g=0 -> mt=0)
  v16h cur[4];
  load_gate_frags(Wih, Whh, 0, r, hi, cur);

#pragma unroll
  for (int k = 0; k < 4; ++k) {  // chunk of hidden units: 16k..16k+15
    v8f acc[2][4];               // [tile][gate]
#pragma unroll
    for (int g = 0; g < 4; ++g) {
      const int idx = k * 4 + g;
      const int mt  = k + 4 * g;  // row tile of g^T (gate-major)
      // prefetch next gate tile's fragments before consuming this one's
      v16h nxt[4];
      if (idx < 15) {
        const int ni = idx + 1;
        const int nmt = (ni >> 2) + 4 * (ni & 3);
        load_gate_frags(Wih, Whh, nmt, r, hi, nxt);
      }
      GF32* bp = bias + mt * 16 + 8 * hi;
      f32x4 b0 = *(GFl4*)bp;
      f32x4 b1 = *((GFl4*)bp + 1);
      v8f bi = (v8f){b0[0], b0[1], b0[2], b0[3], b1[0], b1[1], b1[2], b1[3]};
      acc[0][g] = bi; acc[1][g] = bi;

      acc[0][g] = wmma16(cur[0], xf[0][0], acc[0][g]);
      acc[1][g] = wmma16(cur[0], xf[1][0], acc[1][g]);
      acc[0][g] = wmma16(cur[1], xf[0][1], acc[0][g]);
      acc[1][g] = wmma16(cur[1], xf[1][1], acc[1][g]);
      acc[0][g] = wmma16(cur[2], st.hf[0][0], acc[0][g]);
      acc[1][g] = wmma16(cur[2], st.hf[1][0], acc[1][g]);
      acc[0][g] = wmma16(cur[3], st.hf[0][1], acc[0][g]);
      acc[1][g] = wmma16(cur[3], st.hf[1][1], acc[1][g]);

      if (idx < 15) {
#pragma unroll
        for (int q = 0; q < 4; ++q) cur[q] = nxt[q];
      }
    }
#pragma unroll
    for (int tl = 0; tl < 2; ++tl) {
      float hv[8];
#pragma unroll
      for (int e = 0; e < 8; ++e) {
        float iv = sigmf(acc[tl][0][e]);
        float fv = sigmf(acc[tl][1][e]);
        float gv = htanh(acc[tl][2][e]);
        float ov = sigmf(acc[tl][3][e]);
        float cv = fv * st.c[tl][k][e] + iv * gv;
        st.c[tl][k][e] = cv;
        hv[e] = ov * htanh(cv);
      }
#pragma unroll
      for (int d = 0; d < 4; ++d)
        hpk[tl][k][d] = pack2h(hv[2 * d], hv[2 * d + 1]);
    }
  }
  // C/D chunks -> B-fragments: only a lane<->lane^16 exchange needed.
#pragma unroll
  for (int tl = 0; tl < 2; ++tl)
#pragma unroll
    for (int f = 0; f < 2; ++f) {
      Frag nf;
#pragma unroll
      for (int d = 0; d < 4; ++d) {
        unsigned int v0 = hpk[tl][2 * f][d], v1 = hpk[tl][2 * f + 1][d];
        unsigned int s1 = (unsigned int)__shfl_xor((int)v1, 16, 32);
        unsigned int s0 = (unsigned int)__shfl_xor((int)v0, 16, 32);
        nf.u[d]     = (hi == 0) ? v0 : s1;  // K = kb+0..7
        nf.u[4 + d] = (hi == 0) ? s0 : v1;  // K = kb+8..15
      }
      st.hf[tl][f] = nf.h;
    }
}

// dot(h_row[seq], w[0..63]) from one tile's B-fragments; valid in all lanes.
DEVINL float fragdot(const v16h* hf, const float* __restrict__ w, int hi) {
  float p = 0.f;
#pragma unroll
  for (int f = 0; f < 2; ++f) {
    Frag fr; fr.h = hf[f];
    const int kb = 32 * f + 16 * hi;
    const f32x4* wq = (const f32x4*)(w + kb);
    f32x4 w0 = wq[0], w1 = wq[1], w2 = wq[2], w3 = wq[3];
    p = fmaf(h2f_lo(fr.u[0]), w0[0], p); p = fmaf(h2f_hi(fr.u[0]), w0[1], p);
    p = fmaf(h2f_lo(fr.u[1]), w0[2], p); p = fmaf(h2f_hi(fr.u[1]), w0[3], p);
    p = fmaf(h2f_lo(fr.u[2]), w1[0], p); p = fmaf(h2f_hi(fr.u[2]), w1[1], p);
    p = fmaf(h2f_lo(fr.u[3]), w1[2], p); p = fmaf(h2f_hi(fr.u[3]), w1[3], p);
    p = fmaf(h2f_lo(fr.u[4]), w2[0], p); p = fmaf(h2f_hi(fr.u[4]), w2[1], p);
    p = fmaf(h2f_lo(fr.u[5]), w2[2], p); p = fmaf(h2f_hi(fr.u[5]), w2[3], p);
    p = fmaf(h2f_lo(fr.u[6]), w3[0], p); p = fmaf(h2f_hi(fr.u[6]), w3[1], p);
    p = fmaf(h2f_lo(fr.u[7]), w3[2], p); p = fmaf(h2f_hi(fr.u[7]), w3[3], p);
  }
  p += __shfl_xor(p, 16, 32);
  return p;
}

// per-lane embedding weights (features this lane owns in the B layout)
struct EmbReg { float w0[32], w1[32], b[32]; };

DEVINL void embed_frags(float px, float py, const EmbReg& er, v16h* xf) {
#pragma unroll
  for (int f = 0; f < 2; ++f) {
    Frag fr;
#pragma unroll
    for (int d = 0; d < 8; ++d) {
      float a = fmaf(px, er.w0[f * 16 + 2 * d],
                     fmaf(py, er.w1[f * 16 + 2 * d], er.b[f * 16 + 2 * d]));
      float b = fmaf(px, er.w0[f * 16 + 2 * d + 1],
                     fmaf(py, er.w1[f * 16 + 2 * d + 1], er.b[f * 16 + 2 * d + 1]));
      fr.u[d] = pack2h(a, b);
    }
    xf[f] = fr.h;
  }
}

// ---------------- kernels ----------------

__global__ __launch_bounds__(256) void convert_weights_kernel(
    const float* __restrict__ w0, const float* __restrict__ w1,
    const float* __restrict__ w2, const float* __restrict__ w3,
    const float* __restrict__ w4, const float* __restrict__ w5,
    const float* __restrict__ w6, const float* __restrict__ w7,
    _Float16* __restrict__ outw) {
  int i = blockIdx.x * blockDim.x + threadIdx.x;
  if (i >= 8 * 16384) return;
  int m = i >> 14, j = i & 16383;
  const float* s;
  switch (m) {
    case 0: s = w0; break; case 1: s = w1; break;
    case 2: s = w2; break; case 3: s = w3; break;
    case 4: s = w4; break; case 5: s = w5; break;
    case 6: s = w6; break; default: s = w7; break;
  }
  outw[i] = (_Float16)s[j];
}

#define ADJ_OFF 0
#define DEC_OFF 786432
#define CM_OFF  1572864
#define CV_OFF  1671168

__global__ __launch_bounds__(256) void mtapfm_kernel(
    const float* __restrict__ hist, const float* __restrict__ goal,
    const float* __restrict__ emb_w, const float* __restrict__ emb_b,
    const float* __restrict__ bE0, const float* __restrict__ bE1,
    const float* __restrict__ bD0, const float* __restrict__ bD1,
    const float* __restrict__ coeff_w, const float* __restrict__ coeff_b,
    const float* __restrict__ out_w, const float* __restrict__ out_b,
    const _Float16* __restrict__ wts, float* __restrict__ out) {
  const int lane    = threadIdx.x & 31;
  const int wave    = threadIdx.x >> 5;
  const int hi      = lane >> 4;
  const int r       = lane & 15;
  const int seqbase = (blockIdx.x * 8 + wave) * 32;  // 2 tiles of 16 seqs
  int seqs[2];
  seqs[0] = seqbase + r;
  seqs[1] = seqbase + 16 + r;
  const int group0 = seqbase >> 4;  // (b*A + a) for tile 0; tile1 = group0+1

  // embedding weights, per-lane register resident (shared by both tiles)
  EmbReg er;
#pragma unroll
  for (int f = 0; f < 2; ++f)
#pragma unroll
    for (int j = 0; j < 16; ++j) {
      int feat = 32 * f + 16 * hi + j;
      er.w0[f * 16 + j] = emb_w[feat * 2 + 0];
      er.w1[f * 16 + j] = emb_w[feat * 2 + 1];
      er.b[f * 16 + j]  = emb_b[feat];
    }

  // last nonzero position in history (per tile)
  float lx[2] = {0.f, 0.f}, ly[2] = {0.f, 0.f};
#pragma unroll 1
  for (int t = 0; t < 20; ++t) {
#pragma unroll
    for (int tl = 0; tl < 2; ++tl) {
      float x = hist[(seqs[tl] * 20 + t) * 2 + 0];
      float y = hist[(seqs[tl] * 20 + t) * 2 + 1];
      if (x != 0.f || y != 0.f) { lx[tl] = x; ly[tl] = y; }
    }
  }

  LayerState2 s0, s1;
  {
    Frag zf;
#pragma unroll
    for (int i = 0; i < 8; ++i) zf.u[i] = 0u;
    v8f zc = {0.f, 0.f, 0.f, 0.f, 0.f, 0.f, 0.f, 0.f};
#pragma unroll
    for (int tl = 0; tl < 2; ++tl) {
#pragma unroll
      for (int f = 0; f < 2; ++f) { s0.hf[tl][f] = zf.h; s1.hf[tl][f] = zf.h; }
#pragma unroll
      for (int k = 0; k < 4; ++k) { s0.c[tl][k] = zc; s1.c[tl][k] = zc; }
    }
  }

  const _Float16* wE0i = wts;           const _Float16* wE0h = wts + 16384;
  const _Float16* wE1i = wts + 32768;   const _Float16* wE1h = wts + 49152;
  const _Float16* wD0i = wts + 65536;   const _Float16* wD0h = wts + 81920;
  const _Float16* wD1i = wts + 98304;   const _Float16* wD1h = wts + 114688;

  // -------- encoder: 20 steps, 2 layers, 2 tiles --------
#pragma unroll 1
  for (int t = 0; t < 20; ++t) {
    v16h xf[2][2];
#pragma unroll
    for (int tl = 0; tl < 2; ++tl) {
      float px = hist[(seqs[tl] * 20 + t) * 2 + 0];
      float py = hist[(seqs[tl] * 20 + t) * 2 + 1];
      embed_frags(px, py, er, xf[tl]);
    }
    lstm_cell2(wE0i, wE0h, bE0, xf, s0, lane);
    lstm_cell2(wE1i, wE1h, bE1, s0.hf, s1, lane);
  }

  // -------- coeffs from top encoder state --------
  float k1[2], k2[2], kr[2];
#pragma unroll
  for (int tl = 0; tl < 2; ++tl) {
    k1[tl] = fragdot(s1.hf[tl], coeff_w + 0,   hi) + coeff_b[0];
    k2[tl] = fragdot(s1.hf[tl], coeff_w + 64,  hi) + coeff_b[1];
    kr[tl] = fragdot(s1.hf[tl], coeff_w + 128, hi) + coeff_b[2];
    if (lane < 16) {
      int s = seqs[tl];
      out[CM_OFF + s * 3 + 0] = k1[tl];
      out[CM_OFF + s * 3 + 1] = k2[tl];
      out[CM_OFF + s * 3 + 2] = kr[tl];
      out[CV_OFF + s * 3 + 0] = 0.f;
      out[CV_OFF + s * 3 + 1] = 0.f;
      out[CV_OFF + s * 3 + 2] = 0.f;
    }
  }

  // -------- fused decoder + social-force adjustment --------
  float gx[2], gy[2];
  gx[0] = goal[group0 * 2 + 0];       gy[0] = goal[group0 * 2 + 1];
  gx[1] = goal[(group0 + 1) * 2 + 0]; gy[1] = goal[(group0 + 1) * 2 + 1];
  const float ob0 = out_b[0], ob1 = out_b[1];
  float inx[2], iny[2], prevx[2], prevy[2], curx[2], cury[2];
#pragma unroll
  for (int tl = 0; tl < 2; ++tl) {
    inx[tl] = lx[tl]; iny[tl] = ly[tl];
    prevx[tl] = lx[tl]; prevy[tl] = ly[tl];
    curx[tl] = lx[tl]; cury[tl] = ly[tl];
  }

#pragma unroll 1
  for (int t = 0; t < 12; ++t) {
    v16h xf[2][2];
#pragma unroll
    for (int tl = 0; tl < 2; ++tl) embed_frags(inx[tl], iny[tl], er, xf[tl]);
    lstm_cell2(wD0i, wD0h, bD0, xf, s0, lane);
    lstm_cell2(wD1i, wD1h, bD1, s0.hf, s1, lane);
#pragma unroll
    for (int tl = 0; tl < 2; ++tl) {
      float ox = fragdot(s1.hf[tl], out_w + 0,  hi) + ob0;
      float oy = fragdot(s1.hf[tl], out_w + 64, hi) + ob1;
      float predx = prevx[tl] + ox, predy = prevy[tl] + oy;
      if (lane < 16) {
        out[DEC_OFF + seqs[tl] * 24 + t * 2 + 0] = predx;
        out[DEC_OFF + seqs[tl] * 24 + t * 2 + 1] = predy;
      }
      prevx[tl] = predx; prevy[tl] = predy;
      inx[tl] = predx;   iny[tl] = predy;

      // social forces among the 16 entities of this tile's group
      float fgx = k1[tl] * (gx[tl] - curx[tl]);
      float fgy = k1[tl] * (gy[tl] - cury[tl]);
      float fpx = k2[tl] * (predx - curx[tl]);
      float fpy = k2[tl] * (predy - cury[tl]);
      float frx = 0.f, fry = 0.f;
#pragma unroll
      for (int j = 0; j < 16; ++j) {
        float cjx = __shfl(curx[tl], j, 32);
        float cjy = __shfl(cury[tl], j, 32);
        float dx = curx[tl] - cjx, dy = cury[tl] - cjy;
        float dist = sqrtf(dx * dx + dy * dy) + 1e-6f;
        float m = ((dist < 0.5f) && (dist > 1e-5f) && (j != r)) ? 1.0f : 0.0f;
        float s = m * kr[tl] * hrcp(dist * dist);
        frx = fmaf(s, dx, frx);
        fry = fmaf(s, dy, fry);
      }
      float nx = curx[tl] + fgx + fpx + frx;
      float ny = cury[tl] + fgy + fpy + fry;
      if (t > 0) {
        float dxp = nx - curx[tl], dyp = ny - cury[tl];
        float sp = sqrtf(dxp * dxp + dyp * dyp) * 10.0f;  // /DT, DT=0.1
        float sc = fminf(fmaxf(sp, 3.47395f), 4.70005f) / fmaxf(sp, 1e-8f);
        nx = curx[tl] + dxp * sc;
        ny = cury[tl] + dyp * sc;
      }
      if (lane < 16) {
        out[ADJ_OFF + seqs[tl] * 24 + t * 2 + 0] = nx;
        out[ADJ_OFF + seqs[tl] * 24 + t * 2 + 1] = ny;
      }
      curx[tl] = nx; cury[tl] = ny;
    }
  }
}

// ---------------- host ----------------

extern "C" void kernel_launch(void* const* d_in, const int* in_sizes, int n_in,
                              void* d_out, int out_size, void* d_ws, size_t ws_size,
                              hipStream_t stream) {
  // Top-level: dict order should give hist, goal; tolerate sorted order too.
  int ih = 0, ig = 1;
  if (n_in >= 2 && in_sizes[0] == 4096 && in_sizes[1] != 4096) { ig = 0; ih = 1; }
  const float* hist = (const float*)d_in[ih];
  const float* goal = (const float*)d_in[ig];

  const float *emb_w, *emb_b, *cw, *cb, *ow, *ob;
  const float *Wih[4], *Whh[4], *bias[4];  // enc0, enc1, dec0, dec1

  if (n_in >= 20 && in_sizes[2] == 3) {
    // JAX pytree (sorted-key) order: coeff_b, coeff_w, dec[0]{Whh,Wih,b},
    // dec[1]{...}, emb_b, emb_w, enc[0]{Whh,Wih,b}, enc[1]{...}, out_b, out_w
    cb = (const float*)d_in[2];  cw = (const float*)d_in[3];
    Whh[2] = (const float*)d_in[4];  Wih[2] = (const float*)d_in[5];  bias[2] = (const float*)d_in[6];
    Whh[3] = (const float*)d_in[7];  Wih[3] = (const float*)d_in[8];  bias[3] = (const float*)d_in[9];
    emb_b = (const float*)d_in[10]; emb_w = (const float*)d_in[11];
    Whh[0] = (const float*)d_in[12]; Wih[0] = (const float*)d_in[13]; bias[0] = (const float*)d_in[14];
    Whh[1] = (const float*)d_in[15]; Wih[1] = (const float*)d_in[16]; bias[1] = (const float*)d_in[17];
    ob = (const float*)d_in[18]; ow = (const float*)d_in[19];
  } else {
    // insertion order: emb_w, emb_b, enc[0]{Wih,Whh,b}, enc[1], dec[0], dec[1],
    // coeff_w, coeff_b, out_w, out_b
    emb_w = (const float*)d_in[2]; emb_b = (const float*)d_in[3];
    Wih[0] = (const float*)d_in[4];  Whh[0] = (const float*)d_in[5];  bias[0] = (const float*)d_in[6];
    Wih[1] = (const float*)d_in[7];  Whh[1] = (const float*)d_in[8];  bias[1] = (const float*)d_in[9];
    Wih[2] = (const float*)d_in[10]; Whh[2] = (const float*)d_in[11]; bias[2] = (const float*)d_in[12];
    Wih[3] = (const float*)d_in[13]; Whh[3] = (const float*)d_in[14]; bias[3] = (const float*)d_in[15];
    cw = (const float*)d_in[16]; cb = (const float*)d_in[17];
    ow = (const float*)d_in[18]; ob = (const float*)d_in[19];
  }

  _Float16* wsH = (_Float16*)d_ws;  // 8 matrices x 16384 f16 = 256 KB
  convert_weights_kernel<<<512, 256, 0, stream>>>(
      Wih[0], Whh[0], Wih[1], Whh[1], Wih[2], Whh[2], Wih[3], Whh[3], wsH);

  // 32768 seqs / (8 waves * 32 seqs) = 128 blocks
  mtapfm_kernel<<<128, 256, 0, stream>>>(
      hist, goal, emb_w, emb_b, bias[0], bias[1], bias[2], bias[3],
      cw, cb, ow, ob, wsH, (float*)d_out);
}
